// GraphConditionEncoder_67680094650551
// MI455X (gfx1250) — matrix operand order repaired and checked by reading, not compiled, not run on hardware
//
#include <hip/hip_runtime.h>
#include <hip/hip_bf16.h>

// ---------------- problem constants (match reference) ----------------
#define Nn   30000
#define Ee   240000
#define Bb   64
#define Ff   64
#define Hh   4
#define HF   256        // H*F
#define E2   (Ee + Nn)  // edges + self loops
#define GCOL 832        // F*(1+H*T)
#define BOT  256

typedef __attribute__((ext_vector_type(16))) _Float16 v16h;
typedef __attribute__((ext_vector_type(8)))  _Float16 v8h;
typedef __attribute__((ext_vector_type(8)))  float    v8f;

union AV { v16h v; v8h h[2]; };

// order-preserving float<->uint encode for atomic max over signed floats
__device__ __forceinline__ unsigned fenc(float f) {
    unsigned b = __float_as_uint(f);
    return b ^ (unsigned)(((int)b >> 31) | 0x80000000);
}
__device__ __forceinline__ float fdec(unsigned u) {
    unsigned b = (u & 0x80000000u) ? (u ^ 0x80000000u) : ~u;
    return __uint_as_float(b);
}

// ---------------- utility ----------------
__global__ void k_fill_u32(unsigned* __restrict__ p, unsigned v, int n) {
    int t = blockIdx.x * blockDim.x + threadIdx.x;
    if (t < n) p[t] = v;
}

// out[n*256+j] = bias[j]
__global__ void k_bias(float* __restrict__ out, const float* __restrict__ bias) {
    int t = blockIdx.x * blockDim.x + threadIdx.x;   // N*256 threads
    out[t] = bias[t & 255];
}

// ---------------- node embedding: x0 = relu(sf @ bbox_w + bbox_b) ----------------
// writes f16 xh (stride 64) and pools g[:, 0:64] via raw-bit atomicMax (values >= 0)
__global__ void k_embed(const float* __restrict__ sf, const float* __restrict__ w,
                        const float* __restrict__ b, const int* __restrict__ batch,
                        _Float16* __restrict__ xh, unsigned* __restrict__ g) {
    int t = blockIdx.x * blockDim.x + threadIdx.x;   // N*64 threads
    int n = t >> 6, f = t & 63;
    float acc = b[f];
    #pragma unroll
    for (int i = 0; i < 5; ++i) acc += sf[n * 5 + i] * w[i * 64 + f];
    float v = acc > 0.f ? acc : 0.f;
    xh[t] = (_Float16)v;
    atomicMax(&g[batch[n] * GCOL + f], __float_as_uint(v));
}

// ---------------- weight transpose + f16: wT[n*K + k] = w[k*256 + n] ----------------
__global__ void k_wT(const float* __restrict__ w, _Float16* __restrict__ wT, int K) {
    int t = blockIdx.x * blockDim.x + threadIdx.x;   // K*256 threads
    int k = t >> 8, n = t & 255;
    wT[n * K + k] = (_Float16)w[t];
}

// ---------------- WMMA GEMM: xs[M,256] = xh[M,K] @ W[K,256]  (W given as wT[256,K]) ----
// block: 128 thr = 4 waves. Each wave owns a 16(M) x 64(N) strip: one A-tile load is
// reused across 4 WMMAs per K-step (4 accumulators). 4 waves cover all 256 columns.
// grid: (M/16).
__global__ void k_gemm_wmma(const _Float16* __restrict__ A,   // [M,K] row-major f16
                            const _Float16* __restrict__ BT,  // [256,K] = W^T f16
                            float* __restrict__ C,            // [M,256] f32
                            int K) {
    const int lane    = threadIdx.x & 31;
    const int wave    = threadIdx.x >> 5;
    const int hi      = lane >> 4;          // 0: lanes 0-15, 1: lanes 16-31
    const int rc      = lane & 15;
    const int row     = blockIdx.x * 16 + rc;
    const int colBase = wave * 64;          // this wave's 64-column strip

    const _Float16* aRow = A  + (size_t)row * K;
    const _Float16* bRow = BT + (size_t)(colBase + rc) * K;   // tile tt at +tt*16*K

    v8f acc0 = {}, acc1 = {}, acc2 = {}, acc3 = {};
    for (int kb = 0; kb < K; kb += 32) {
        // A 16x32 f16 per-lane layout: 16 contiguous K values per lane (two 8-half chunks)
        AV a;
        a.h[0] = *(const v8h*)(aRow + kb + hi * 16);
        a.h[1] = *(const v8h*)(aRow + kb + hi * 16 + 8);
        // B 32x16 f16 per-lane layout: 16 contiguous K values per lane
        const _Float16* bp = bRow + kb + hi * 16;
        v16h b0 = *(const v16h*)(bp);
        v16h b1 = *(const v16h*)(bp + (size_t)16 * K);
        v16h b2 = *(const v16h*)(bp + (size_t)32 * K);
        v16h b3 = *(const v16h*)(bp + (size_t)48 * K);
        acc0 = __builtin_amdgcn_wmma_f32_16x16x32_f16(false, a.v, false, b0, (short)0, acc0, false, false);
        acc1 = __builtin_amdgcn_wmma_f32_16x16x32_f16(false, a.v, false, b1, (short)0, acc1, false, false);
        acc2 = __builtin_amdgcn_wmma_f32_16x16x32_f16(false, a.v, false, b2, (short)0, acc2, false, false);
        acc3 = __builtin_amdgcn_wmma_f32_16x16x32_f16(false, a.v, false, b3, (short)0, acc3, false, false);
    }
    // D layout: VGPR r -> row r + 8*hi, col = lane&15
    const int mBase = blockIdx.x * 16 + hi * 8;
    float* cp = C + (size_t)mBase * 256 + colBase + rc;
    #pragma unroll
    for (int r = 0; r < 8; ++r) {
        cp[(size_t)r * 256 +  0] = acc0[r];
        cp[(size_t)r * 256 + 16] = acc1[r];
        cp[(size_t)r * 256 + 32] = acc2[r];
        cp[(size_t)r * 256 + 48] = acc3[r];
    }
}

// ---------------- attention coefficients: a_*[n,h] = dot(xs[n,h,:], att_*[h,:]) -------
__global__ void k_attn(const float* __restrict__ xs, const float* __restrict__ att_s,
                       const float* __restrict__ att_d,
                       float* __restrict__ as_, float* __restrict__ ad_) {
    int t = blockIdx.x * blockDim.x + threadIdx.x;   // N*H threads
    if (t >= Nn * Hh) return;
    int n = t >> 2, h = t & 3;
    const float* xp = xs + (size_t)n * HF + h * Ff;
    float s0 = 0.f, s1 = 0.f;
    #pragma unroll 8
    for (int f = 0; f < Ff; ++f) {
        float x = xp[f];
        s0 += x * att_s[h * Ff + f];
        s1 += x * att_d[h * Ff + f];
    }
    as_[t] = s0;
    ad_[t] = s1;
}

__device__ __forceinline__ void edge_sd(const int* __restrict__ ei, int e, int& s, int& d) {
    if (e < Ee) { s = ei[e]; d = ei[Ee + e]; }
    else        { s = e - Ee; d = e - Ee; }      // self loop
}

__device__ __forceinline__ float edge_logit(const float* as_, const float* ad_,
                                            int s, int d, int h) {
    float v = as_[s * Hh + h] + ad_[d * Hh + h];
    return v > 0.f ? v : 0.2f * v;               // leaky_relu(0.2)
}

// ---------------- edge pass 1: segment max over dst ----------------
__global__ void k_edge_max(const int* __restrict__ ei, const float* __restrict__ as_,
                           const float* __restrict__ ad_, unsigned* __restrict__ mmax) {
    int t = blockIdx.x * blockDim.x + threadIdx.x;   // E2*H threads
    if (t >= E2 * Hh) return;
    int e = t >> 2, h = t & 3, s, d;
    edge_sd(ei, e, s, d);
    atomicMax(&mmax[d * Hh + h], fenc(edge_logit(as_, ad_, s, d, h)));
}

// ---------------- edge pass 2: denom = segment sum of exp(l - m) ----------------
__global__ void k_edge_den(const int* __restrict__ ei, const float* __restrict__ as_,
                           const float* __restrict__ ad_, const unsigned* __restrict__ mmax,
                           float* __restrict__ denom) {
    int t = blockIdx.x * blockDim.x + threadIdx.x;   // E2*H threads
    if (t >= E2 * Hh) return;
    int e = t >> 2, h = t & 3, s, d;
    edge_sd(ei, e, s, d);
    float ex = __expf(edge_logit(as_, ad_, s, d, h) - fdec(mmax[d * Hh + h]));
    atomicAdd(&denom[d * Hh + h], ex);
}

// ---------------- edge pass 3: out[d,:] += alpha * xs[s,:] ----------------
// thread t = e*256 + j  (j = h*64 + f): coalesced xs read + coalesced atomic scatter
__global__ void k_edge_msg(const int* __restrict__ ei, const float* __restrict__ as_,
                           const float* __restrict__ ad_, const unsigned* __restrict__ mmax,
                           const float* __restrict__ denom, const float* __restrict__ xs,
                           float* __restrict__ out) {
    int t = blockIdx.x * blockDim.x + threadIdx.x;   // E2*256 threads exactly
    int e = t >> 8, j = t & 255, h = j >> 6, s, d;
    edge_sd(ei, e, s, d);
    float ex    = __expf(edge_logit(as_, ad_, s, d, h) - fdec(mmax[d * Hh + h]));
    float alpha = ex / (denom[d * Hh + h] + 1e-16f);
    atomicAdd(&out[(size_t)d * HF + j], alpha * xs[(size_t)s * HF + j]);
}

// ---------------- relu + f16 convert (next layer input) + batch max-pool -------------
__global__ void k_relu_pool(const float* __restrict__ out, const int* __restrict__ batch,
                            _Float16* __restrict__ xh, unsigned* __restrict__ g, int gcol) {
    int t = blockIdx.x * blockDim.x + threadIdx.x;   // N*256 threads
    int n = t >> 8, j = t & 255;
    float v = out[t];
    v = v > 0.f ? v : 0.f;
    xh[t] = (_Float16)v;
    atomicMax(&g[batch[n] * GCOL + gcol + j], __float_as_uint(v));
}

// ---------------- final: y = g_embed @ agg_w + agg_b  ([64,832]x[832,256]) -----------
__global__ void k_agg(const unsigned* __restrict__ g, const float* __restrict__ w,
                      const float* __restrict__ b, float* __restrict__ y) {
    int bi = blockIdx.x;          // 64 blocks
    int o  = threadIdx.x;         // 256 threads
    const float* gp = (const float*)(g + bi * GCOL);   // raw bits == nonneg floats
    float acc = b[o];
    for (int c = 0; c < GCOL; ++c) acc += gp[c] * w[c * BOT + o];
    y[bi * BOT + o] = acc;
}

// =====================================================================================
extern "C" void kernel_launch(void* const* d_in, const int* in_sizes, int n_in,
                              void* d_out, int out_size, void* d_ws, size_t ws_size,
                              hipStream_t stream) {
    (void)in_sizes; (void)n_in; (void)out_size; (void)ws_size;

    const float* sf      = (const float*)d_in[0];
    const int*   ei      = (const int*)  d_in[1];
    const int*   batch   = (const int*)  d_in[2];
    const float* bbox_w  = (const float*)d_in[3];
    const float* bbox_b  = (const float*)d_in[4];
    const float* agg_w   = (const float*)d_in[17];
    const float* agg_b   = (const float*)d_in[18];
    float*       y       = (float*)d_out;

    // ---- workspace carve-up (all offsets 256B aligned) ----
    char* ws = (char*)d_ws;
    _Float16* xh    = (_Float16*)(ws);                         // N*256 f16 = 15,360,000
    _Float16* wT    = (_Float16*)(ws + 15360000);              // 256*256 f16 = 131,072
    float*    xs    = (float*)   (ws + 15491072);              // N*256 f32 = 30,720,000
    float*    out   = (float*)   (ws + 46211072);              // N*256 f32 = 30,720,000
    float*    a_src = (float*)   (ws + 76931072);              // N*4  f32  = 480,000
    float*    a_dst = (float*)   (ws + 77411072);              // N*4  f32  = 480,000
    unsigned* mmax  = (unsigned*)(ws + 77891072);              // N*4  u32  = 480,000
    float*    denom = (float*)   (ws + 78371072);              // N*4  f32  = 480,000
    unsigned* g     = (unsigned*)(ws + 78851072);              // B*832 u32 = 212,992

    // ---- pooled embedding buffer: init to 0 (valid: all pooled values >= 0) ----
    k_fill_u32<<<(Bb * GCOL + 255) / 256, 256, 0, stream>>>(g, 0u, Bb * GCOL);

    // ---- node embedding (K=5 scalar) + g0 pool ----
    k_embed<<<(Nn * Ff) / 256, 256, 0, stream>>>(sf, bbox_w, bbox_b, batch, xh, g);

    // ---- 3 GAT layers ----
    for (int L = 0; L < 3; ++L) {
        const int K = (L == 0) ? Ff : HF;
        const float* lw   = (const float*)d_in[5 + 4 * L];
        const float* asc  = (const float*)d_in[6 + 4 * L];
        const float* adc  = (const float*)d_in[7 + 4 * L];
        const float* bias = (const float*)d_in[8 + 4 * L];

        // transpose + f16 weight
        k_wT<<<K, 256, 0, stream>>>(lw, wT, K);

        // xs = xh @ W  via v_wmma_f32_16x16x32_f16 (4 wmma / K-step / wave)
        k_gemm_wmma<<<Nn / 16, 128, 0, stream>>>(xh, wT, xs, K);

        // attention coefficients
        k_attn<<<(Nn * Hh + 255) / 256, 256, 0, stream>>>(xs, asc, adc, a_src, a_dst);

        // segment softmax state
        k_fill_u32<<<(Nn * Hh + 255) / 256, 256, 0, stream>>>(mmax, 0u, Nn * Hh);
        k_fill_u32<<<(Nn * Hh + 255) / 256, 256, 0, stream>>>((unsigned*)denom, 0u, Nn * Hh);

        k_edge_max<<<(E2 * Hh + 255) / 256, 256, 0, stream>>>(ei, a_src, a_dst, mmax);
        k_edge_den<<<(E2 * Hh + 255) / 256, 256, 0, stream>>>(ei, a_src, a_dst, mmax, denom);

        // out = bias, then scatter messages
        k_bias<<<(Nn * HF) / 256, 256, 0, stream>>>(out, bias);
        k_edge_msg<<<E2, 256, 0, stream>>>(ei, a_src, a_dst, mmax, denom, xs, out);

        // relu, convert to f16 for next layer, pool g[:, 64 + L*256 ...]
        k_relu_pool<<<(Nn * HF) / 256, 256, 0, stream>>>(out, batch, xh, g, Ff + L * HF);
    }

    // ---- final dense: y = g_embed @ agg_w + agg_b ----
    k_agg<<<Bb, BOT, 0, stream>>>(g, agg_w, agg_b, y);
}
